// Mambaformer_48361331753424
// MI455X (gfx1250) — compile-verified, matrix operand order checked by reading, not run on hardware
//
#include <hip/hip_runtime.h>
#include <math.h>

// ---------------------------------------------------------------------------
// MI455X / gfx1250 implementation of the "Mambaformer" reference.
// Matrix ops use V_WMMA_F32_16X16X4_F32 (full fp32 — the model is bandwidth
// bound on MI455X, so fp32 WMMA is free and keeps reference numerics).
// Fallback (if the f32 builtin is absent) uses the codegen-confirmed
// v_wmma_f32_16x16x32_f16 with the same LDS tiling.
// ---------------------------------------------------------------------------

typedef __attribute__((ext_vector_type(2)))  float    v2f;
typedef __attribute__((ext_vector_type(8)))  float    v8f;
typedef __attribute__((ext_vector_type(16))) _Float16 v16h;

#define HAS_F32WMMA __has_builtin(__builtin_amdgcn_wmma_f32_16x16x4_f32)

namespace {
constexpr int kB = 8, kS = 256, kD = 512, kH = 8;
constexpr int kDI = 1024, kDSt = 16, kDTR = 32;
constexpr int kR = kB * kS;   // 2048 rows (B*S)
}

// ---------------------------------------------------------------------------
// Generic batched GEMM:  C = act(A @ op(B) + bias) + res
//   A: (M,K) row-major, lda.   op(B): transB==0 -> B is (N,K) weights (B^T
//   applied);  transB==1 -> B is (K,N).   Batch z decomposed as
//   (z/inner, z%inner) with independent (outer,inner) strides per operand.
// Tile: 64x64 per 128-thread block (4 waves), wave = 2x2 of 16x16 WMMA tiles.
// Requires M%64==0, N%64==0, K%32==0 (true for every GEMM in this model).
// WMMA f32 16x16x4 operand layout (ISA 7.12.2):
//   A frag (16x4): lanes 0-15 rows, VGPR0/1 = K {0,1}; lanes 16-31 K {2,3}.
//   B frag (4x16): lanes 0-15 cols with K {0,1}; lanes 16-31 K {2,3}.
//   C/D: VGPR e -> row e (lanes<16) / row e+8 (lanes>=16), col = lane%16.
// ---------------------------------------------------------------------------
__global__ __launch_bounds__(128)
void gemm_wmma_kernel(const float* __restrict__ A, long lda, long aO, long aI,
                      const float* __restrict__ Bw, long ldb, long bO, long bI,
                      int transB,
                      const float* __restrict__ bias,
                      const float* __restrict__ res,
                      float* __restrict__ C, long ldc, long cO, long cI,
                      int M, int N, int K, int innerCnt, int act)
{
  __shared__ float ldsA[64 * 33];   // padded: 33 stride kills bank conflicts
  __shared__ float ldsB[32 * 68];   // [k][n], 68 stride

  const int  z  = blockIdx.z;
  const long zo = (long)(z / innerCnt), zi = (long)(z % innerCnt);
  A  += aO * zo + aI * zi;
  Bw += bO * zo + bI * zi;
  C  += cO * zo + cI * zi;
  if (res) res += cO * zo + cI * zi;

  const int m0 = blockIdx.y * 64, n0 = blockIdx.x * 64;
  const int tid = threadIdx.x, lane = tid & 31, wave = tid >> 5;
  const int wm = (wave >> 1) * 32, wn = (wave & 1) * 32;
  const int l15 = lane & 15, lhi = lane >> 4;

  v8f acc00 = {}, acc01 = {}, acc10 = {}, acc11 = {};

  for (int k0 = 0; k0 < K; k0 += 32) {
    __syncthreads();
    #pragma unroll
    for (int i = tid; i < 64 * 32; i += 128) {
      int r = i >> 5, c = i & 31;
      ldsA[r * 33 + c] = A[(long)(m0 + r) * lda + (k0 + c)];
    }
    if (!transB) {
      #pragma unroll
      for (int i = tid; i < 64 * 32; i += 128) {
        int n = i >> 5, k = i & 31;
        ldsB[k * 68 + n] = Bw[(long)(n0 + n) * ldb + (k0 + k)];
      }
    } else {
      #pragma unroll
      for (int i = tid; i < 64 * 32; i += 128) {
        int k = i >> 6, n = i & 63;
        ldsB[k * 68 + n] = Bw[(long)(k0 + k) * ldb + (n0 + n)];
      }
    }
    if (k0 + 32 < K) {   // lowers to global_prefetch_b8
      __builtin_prefetch(A + (long)(m0 + (tid & 63)) * lda + (k0 + 32), 0, 1);
      __builtin_prefetch(Bw + (transB ? ((long)(k0 + 32) * ldb + n0 + (tid & 63))
                                      : ((long)(n0 + (tid & 63)) * ldb + (k0 + 32))),
                         0, 1);
    }
    __syncthreads();

#if HAS_F32WMMA
    #pragma unroll
    for (int ks = 0; ks < 32; ks += 4) {
      const int kb  = ks + lhi * 2;
      const int ra0 = (wm + l15) * 33, ra1 = (wm + 16 + l15) * 33;
      v2f a0, a1, b0, b1;
      a0.x = ldsA[ra0 + kb];  a0.y = ldsA[ra0 + kb + 1];
      a1.x = ldsA[ra1 + kb];  a1.y = ldsA[ra1 + kb + 1];
      b0.x = ldsB[kb * 68 + wn + l15];      b0.y = ldsB[(kb + 1) * 68 + wn + l15];
      b1.x = ldsB[kb * 68 + wn + 16 + l15]; b1.y = ldsB[(kb + 1) * 68 + wn + 16 + l15];
      acc00 = __builtin_amdgcn_wmma_f32_16x16x4_f32(false, a0, false, b0, (short)0, acc00, false, false);
      acc01 = __builtin_amdgcn_wmma_f32_16x16x4_f32(false, a0, false, b1, (short)0, acc01, false, false);
      acc10 = __builtin_amdgcn_wmma_f32_16x16x4_f32(false, a1, false, b0, (short)0, acc10, false, false);
      acc11 = __builtin_amdgcn_wmma_f32_16x16x4_f32(false, a1, false, b1, (short)0, acc11, false, false);
    }
#else
    // f16 fallback: one 16x16x32 WMMA per tile per K-chunk.
    // 16-bit A 16x32 layout: v16h elem e: g=e>>1,h=e&1;
    //   K = (g<4 ? g*2+h : 16+(g-4)*2+h) + (lane>=16 ? 8 : 0)
    {
      v16h fa0, fa1, fb0, fb1;
      #pragma unroll
      for (int e = 0; e < 16; ++e) {
        int g = e >> 1, h = e & 1;
        int kk = (g < 4 ? g * 2 + h : 16 + (g - 4) * 2 + h) + lhi * 8;
        fa0[e] = (_Float16)ldsA[(wm + l15) * 33 + kk];
        fa1[e] = (_Float16)ldsA[(wm + 16 + l15) * 33 + kk];
        fb0[e] = (_Float16)ldsB[kk * 68 + wn + l15];
        fb1[e] = (_Float16)ldsB[kk * 68 + wn + 16 + l15];
      }
      acc00 = __builtin_amdgcn_wmma_f32_16x16x32_f16(false, fa0, false, fb0, (short)0, acc00, false, false);
      acc01 = __builtin_amdgcn_wmma_f32_16x16x32_f16(false, fa0, false, fb1, (short)0, acc01, false, false);
      acc10 = __builtin_amdgcn_wmma_f32_16x16x32_f16(false, fa1, false, fb0, (short)0, acc10, false, false);
      acc11 = __builtin_amdgcn_wmma_f32_16x16x32_f16(false, fa1, false, fb1, (short)0, acc11, false, false);
    }
#endif
  }

  const int rowAdd = lhi * 8;
  #define EPILOGUE(ACC, MT, NT)                                               \
    _Pragma("unroll")                                                         \
    for (int e = 0; e < 8; ++e) {                                             \
      int gm = m0 + wm + (MT) * 16 + e + rowAdd;                              \
      int gn = n0 + wn + (NT) * 16 + l15;                                     \
      float val = (ACC)[e];                                                   \
      if (bias) val += bias[gn];                                              \
      if (act == 1) val = fmaxf(val, 0.f);                                    \
      else if (act == 2) val = (val > 20.f) ? val : log1pf(expf(val));        \
      if (res) val += res[(long)gm * ldc + gn];                               \
      C[(long)gm * ldc + gn] = val;                                           \
    }
  EPILOGUE(acc00, 0, 0)
  EPILOGUE(acc01, 0, 1)
  EPILOGUE(acc10, 1, 0)
  EPILOGUE(acc11, 1, 1)
  #undef EPILOGUE
}

// ---------------------------------------------------------------------------
// Row LayerNorm family over D=512: one block per row, 256 threads x 2 elems.
//   mode 0: temporal PE + AdaLN (unbiased std, eps=1e-6 outside sqrt)
//   mode 1: torch_ln  (biased var, eps=1e-5 inside sqrt)
//   mode 2: custom_ln (unbiased std, eps=1e-6 outside sqrt)
// r1/r2 optional residual inputs (x + r1 + r2 normalized).
// ---------------------------------------------------------------------------
__global__ __launch_bounds__(256)
void ln512_kernel(const float* __restrict__ x, const float* __restrict__ r1,
                  const float* __restrict__ r2, const float* __restrict__ g,
                  const float* __restrict__ bb, float* __restrict__ out,
                  int mode, int Sdim)
{
  __shared__ float sA[256], sB[256];
  const int tid = threadIdx.x;
  const long base = (long)blockIdx.x * 512;
  float v0 = x[base + tid], v1 = x[base + tid + 256];
  if (r1) { v0 += r1[base + tid]; v1 += r1[base + tid + 256]; }
  if (r2) { v0 += r2[base + tid]; v1 += r2[base + tid + 256]; }
  if (mode == 0) {   // temporal positional encoding, pos = batch index
    float pos = (float)(blockIdx.x / Sdim);
    const float c = -logf(10000.f) / 512.f;
    int d0 = tid, d1 = tid + 256;
    float a0 = pos * expf((float)(d0 & ~1) * c);
    float a1 = pos * expf((float)(d1 & ~1) * c);
    v0 += (d0 & 1) ? cosf(a0) : sinf(a0);
    v1 += (d1 & 1) ? cosf(a1) : sinf(a1);
  }
  sA[tid] = v0 + v1;
  sB[tid] = v0 * v0 + v1 * v1;
  __syncthreads();
  for (int o = 128; o > 0; o >>= 1) {
    if (tid < o) { sA[tid] += sA[tid + o]; sB[tid] += sB[tid + o]; }
    __syncthreads();
  }
  const float mean = sA[0] * (1.f / 512.f);
  if (mode == 1) {
    float var = sB[0] * (1.f / 512.f) - mean * mean;
    float inv = rsqrtf(var + 1e-5f);
    out[base + tid]       = (v0 - mean) * inv * g[tid]       + bb[tid];
    out[base + tid + 256] = (v1 - mean) * inv * g[tid + 256] + bb[tid + 256];
  } else {
    float var = (sB[0] - 512.f * mean * mean) * (1.f / 511.f);
    float inv = 1.f / (sqrtf(var) + 1e-6f);
    out[base + tid]       = g[tid]       * ((v0 - mean) * inv) + bb[tid];
    out[base + tid + 256] = g[tid + 256] * ((v1 - mean) * inv) + bb[tid + 256];
  }
}

// ---------------------------------------------------------------------------
// Depthwise causal conv (width 4) over time + bias + SiLU.
//   in: uz (R, 2*DI), channel c taken from first DI cols; out: (R, DI)
// ---------------------------------------------------------------------------
__global__ __launch_bounds__(256)
void conv_silu_kernel(const float* __restrict__ uz, const float* __restrict__ cw,
                      const float* __restrict__ cb, float* __restrict__ out)
{
  long idx = (long)blockIdx.x * 256 + threadIdx.x;   // over R*DI
  int  c  = (int)(idx & (kDI - 1));
  long rl = idx >> 10;                 // = b*S + l
  int  l  = (int)(rl % kS);
  float acc = cb[c];
  #pragma unroll
  for (int k = 0; k < 4; ++k) {
    int lp = l + k - 3;
    if (lp >= 0) acc += uz[(rl + (k - 3)) * (2 * kDI) + c] * cw[c * 4 + k];
  }
  out[rl * kDI + c] = acc / (1.f + expf(-acc));      // SiLU
}

// ---------------------------------------------------------------------------
// Selective-scan: one thread per (b,d) channel, 16 states in registers.
// Fuses  y = scan + D*u  and  t1 = y * silu(z).
// grid = B * (DI/256) = 32 blocks of 256.
// ---------------------------------------------------------------------------
__global__ __launch_bounds__(256)
void scan_kernel(const float* __restrict__ delta, const float* __restrict__ xdbl,
                 const float* __restrict__ u, const float* __restrict__ uz,
                 const float* __restrict__ Alog, const float* __restrict__ Dp,
                 float* __restrict__ out)
{
  __shared__ float sB[16], sC[16];
  const int tid = threadIdx.x;
  const int b = blockIdx.x >> 2;
  const int d = ((blockIdx.x & 3) << 8) + tid;
  float A[16], h[16];
  #pragma unroll
  for (int s = 0; s < 16; ++s) { A[s] = -expf(Alog[d * 16 + s]); h[s] = 0.f; }
  const float dD = Dp[d];
  for (int l = 0; l < kS; ++l) {
    long r = (long)b * kS + l;
    if (tid < 32) {
      float x = xdbl[r * 64 + 32 + tid];
      if (tid < 16) sB[tid] = x; else sC[tid - 16] = x;
    }
    __syncthreads();
    float dl = delta[r * kDI + d];
    float ul = u[r * kDI + d];
    float du = dl * ul;
    float acc = 0.f;
    #pragma unroll
    for (int s = 0; s < 16; ++s) {
      h[s] = expf(dl * A[s]) * h[s] + du * sB[s];
      acc += h[s] * sC[s];
    }
    float zl = uz[r * (2 * kDI) + kDI + d];
    out[r * kDI + d] = (acc + dD * ul) * (zl / (1.f + expf(-zl)));
    __syncthreads();
  }
}

// ---------------------------------------------------------------------------
// Row softmax over 256 with 1/sqrt(hd)=1/8 scale folded in (in place).
// ---------------------------------------------------------------------------
__global__ __launch_bounds__(256)
void softmax256_kernel(float* __restrict__ sc)
{
  __shared__ float sA[256];
  const int tid = threadIdx.x;
  const long base = (long)blockIdx.x * 256;
  float v = sc[base + tid] * 0.125f;
  sA[tid] = v; __syncthreads();
  for (int o = 128; o > 0; o >>= 1) {
    if (tid < o) sA[tid] = fmaxf(sA[tid], sA[tid + o]);
    __syncthreads();
  }
  float m = sA[0]; __syncthreads();
  float e = expf(v - m);
  sA[tid] = e; __syncthreads();
  for (int o = 128; o > 0; o >>= 1) {
    if (tid < o) sA[tid] += sA[tid + o];
    __syncthreads();
  }
  sc[base + tid] = e / sA[0];
}

// ---------------------------------------------------------------------------
extern "C" void kernel_launch(void* const* d_in, const int* in_sizes, int n_in,
                              void* d_out, int out_size, void* d_ws, size_t ws_size,
                              hipStream_t stream)
{
  (void)in_sizes; (void)n_in; (void)out_size; (void)ws_size;
  const float* text    = (const float*)d_in[0];
  const float* video   = (const float*)d_in[1];
  const float* audio   = (const float*)d_in[2];
  const float* emb_w   = (const float*)d_in[3];
  const float* emb_b   = (const float*)d_in[4];
  const float* ada_a   = (const float*)d_in[5];
  const float* ada_b   = (const float*)d_in[6];
  const float* m_in    = (const float*)d_in[7];
  const float* m_cw    = (const float*)d_in[8];
  const float* m_cb    = (const float*)d_in[9];
  const float* m_xp    = (const float*)d_in[10];
  const float* m_dtw   = (const float*)d_in[11];
  const float* m_dtb   = (const float*)d_in[12];
  const float* m_Alog  = (const float*)d_in[13];
  const float* m_Dp    = (const float*)d_in[14];
  const float* m_outw  = (const float*)d_in[15];
  const float* m_lng   = (const float*)d_in[16];
  const float* m_lnb   = (const float*)d_in[17];
  const float* qkv_w   = (const float*)d_in[18];
  const float* qkv_b   = (const float*)d_in[19];
  const float* aout_w  = (const float*)d_in[20];
  const float* aout_b  = (const float*)d_in[21];
  const float* sub_g   = (const float*)d_in[22];
  const float* sub_b   = (const float*)d_in[23];
  const float* f_w1    = (const float*)d_in[24];
  const float* f_b1    = (const float*)d_in[25];
  const float* f_w2    = (const float*)d_in[26];
  const float* f_b2    = (const float*)d_in[27];
  const float* f_lng   = (const float*)d_in[28];
  const float* f_lnb   = (const float*)d_in[29];
  const float* n1_g    = (const float*)d_in[30];
  const float* n1_b    = (const float*)d_in[31];

  // ---- workspace arena (units of MF = R*512 floats = 4 MiB) ----
  float* W = (float*)d_ws;
  const size_t MF = (size_t)kR * 512;
  float* strm[6]; for (int i = 0; i < 6; ++i) strm[i] = W + (size_t)i * MF;
  float* n1    = W + 6 * MF;
  float* n2    = W + 7 * MF;
  float* qb    = W + 8 * MF;
  float* kb    = W + 9 * MF;
  float* vb    = W + 10 * MF;
  float* ob    = W + 11 * MF;
  float* uz    = W + 12 * MF;   // 4*MF; reused as attention scores
  float* scores= uz;
  float* uconv = W + 16 * MF;   // 2*MF; reused as FFN hidden
  float* delta = W + 18 * MF;   // 2*MF
  float* t1    = W + 20 * MF;   // 2*MF
  float* xdbl  = W + 22 * MF;   // R*64 floats

  auto gemm = [&](const float* A, long lda, long aO, long aI,
                  const float* Bm, long ldb, long bO, long bI, int trB,
                  const float* bias, const float* res,
                  float* Cc, long ldc, long cO, long cI,
                  int M, int N, int K, int batch, int inner, int act) {
    dim3 g(N / 64, M / 64, batch);
    gemm_wmma_kernel<<<g, dim3(128), 0, stream>>>(A, lda, aO, aI, Bm, ldb, bO, bI,
                                                  trB, bias, res, Cc, ldc, cO, cI,
                                                  M, N, K, inner, act);
  };
  auto ln = [&](const float* x, const float* r1, const float* r2,
                const float* g, const float* b, float* out, int mode) {
    ln512_kernel<<<kR, 256, 0, stream>>>(x, r1, r2, g, b, out, mode, kS);
  };

  auto mamba_layer = [&](float* x, int i) {
    gemm(x, 512, 0, 0, m_in + (size_t)i * 2048 * 512, 512, 0, 0, 0,
         nullptr, nullptr, uz, 2048, 0, 0, kR, 2048, 512, 1, 1, 0);
    conv_silu_kernel<<<(kR * kDI) / 256, 256, 0, stream>>>(
        uz, m_cw + (size_t)i * kDI * 4, m_cb + (size_t)i * kDI, uconv);
    gemm(uconv, 1024, 0, 0, m_xp + (size_t)i * 64 * 1024, 1024, 0, 0, 0,
         nullptr, nullptr, xdbl, 64, 0, 0, kR, 64, 1024, 1, 1, 0);
    gemm(xdbl, 64, 0, 0, m_dtw + (size_t)i * 1024 * 32, 32, 0, 0, 0,
         m_dtb + (size_t)i * 1024, nullptr, delta, 1024, 0, 0,
         kR, 1024, 32, 1, 1, 2 /*softplus*/);
    scan_kernel<<<kB * (kDI / 256), 256, 0, stream>>>(
        delta, xdbl, uconv, uz, m_Alog + (size_t)i * kDI * kDSt,
        m_Dp + (size_t)i * kDI, t1);
    gemm(t1, 1024, 0, 0, m_outw + (size_t)i * 512 * 1024, 1024, 0, 0, 0,
         nullptr, x /*residual*/, x, 512, 0, 0, kR, 512, 1024, 1, 1, 0);
    ln(x, nullptr, nullptr, m_lng + (size_t)i * 512, m_lnb + (size_t)i * 512, x, 1);
  };

  auto sub = [&](const float* s1, const float* s2, int i, float* out) {
    ln(s1, nullptr, nullptr, sub_g + (size_t)i * 512, sub_b + (size_t)i * 512, n1, 1);
    ln(s2, nullptr, nullptr, sub_g + (size_t)i * 512, sub_b + (size_t)i * 512, n2, 1);
    const float* wq = qkv_w + ((size_t)i * 3 + 0) * 512 * 512;
    const float* wk = qkv_w + ((size_t)i * 3 + 1) * 512 * 512;
    const float* wv = qkv_w + ((size_t)i * 3 + 2) * 512 * 512;
    const float* bq = qkv_b + ((size_t)i * 3 + 0) * 512;
    const float* bk2= qkv_b + ((size_t)i * 3 + 1) * 512;
    const float* bv = qkv_b + ((size_t)i * 3 + 2) * 512;
    gemm(n1, 512, 0, 0, wq, 512, 0, 0, 0, bq, nullptr, qb, 512, 0, 0, kR, 512, 512, 1, 1, 0);
    gemm(n2, 512, 0, 0, wk, 512, 0, 0, 0, bk2, nullptr, kb, 512, 0, 0, kR, 512, 512, 1, 1, 0);
    gemm(n2, 512, 0, 0, wv, 512, 0, 0, 0, bv, nullptr, vb, 512, 0, 0, kR, 512, 512, 1, 1, 0);
    // scores[b,h] = q[b,:,h] @ k[b,:,h]^T  (batch z=(b,h), inner=H)
    gemm(qb, 512, (long)kS * 512, 64,
         kb, 512, (long)kS * 512, 64, 0,
         nullptr, nullptr,
         scores, 256, (long)kH * kS * kS, (long)kS * kS,
         kS, kS, 64, kB * kH, kH, 0);
    softmax256_kernel<<<kB * kH * kS, 256, 0, stream>>>(scores);
    // o[b,:,h] = att @ v[b,:,h]   (v indexed (K=s, N=hd) -> transB=1)
    gemm(scores, 256, (long)kH * kS * kS, (long)kS * kS,
         vb, 512, (long)kS * 512, 64, 1,
         nullptr, nullptr,
         ob, 512, (long)kS * 512, 64,
         kS, 64, kS, kB * kH, kH, 0);
    gemm(ob, 512, 0, 0, aout_w + (size_t)i * 512 * 512, 512, 0, 0, 0,
         aout_b + (size_t)i * 512, s1 /*residual*/, out, 512, 0, 0,
         kR, 512, 512, 1, 1, 0);
  };

  auto ffn = [&](float* x, int i) {
    gemm(x, 512, 0, 0, f_w1 + (size_t)i * 1024 * 512, 512, 0, 0, 0,
         f_b1 + (size_t)i * 1024, nullptr, uconv, 1024, 0, 0,
         kR, 1024, 512, 1, 1, 1 /*relu*/);
    gemm(uconv, 1024, 0, 0, f_w2 + (size_t)i * 512 * 1024, 1024, 0, 0, 0,
         f_b2 + (size_t)i * 512, x /*residual*/, x, 512, 0, 0,
         kR, 512, 1024, 1, 1, 0);
    ln(x, nullptr, nullptr, f_lng + (size_t)i * 512, f_lnb + (size_t)i * 512, x, 2);
  };

  // ---- stems: emb GEMM, then PE + AdaLN ----
  const float* ins[3] = { text, video, audio };
  float* cur[3] = { strm[0], strm[1], strm[2] };
  float* alt[3] = { strm[3], strm[4], strm[5] };
  for (int j = 0; j < 3; ++j) {
    gemm(ins[j], 512, 0, 0, emb_w, 512, 0, 0, 0, emb_b, nullptr,
         cur[j], 512, 0, 0, kR, 512, 512, 1, 1, 0);
    ln(cur[j], nullptr, nullptr, ada_a, ada_b, cur[j], 0 /*PE+adaln*/);
  }

  // ---- block 1: mamba 0-2, self-attn 0-2 ----
  mamba_layer(cur[0], 0); mamba_layer(cur[1], 1); mamba_layer(cur[2], 2);
  sub(cur[0], cur[0], 0, cur[0]);
  sub(cur[1], cur[1], 1, cur[1]);
  sub(cur[2], cur[2], 2, cur[2]);

  // ---- cross attention (needs old values -> write to alt, then swap) ----
  sub(cur[0], cur[2], 3, alt[0]);   // t2 = sub(t, a)
  sub(cur[1], cur[0], 4, alt[1]);   // v2 = sub(v, t)
  sub(cur[2], cur[1], 5, alt[2]);   // a2 = sub(a, v)
  { float* tmp;
    tmp = cur[0]; cur[0] = alt[0]; alt[0] = tmp;
    tmp = cur[1]; cur[1] = alt[1]; alt[1] = tmp;
    tmp = cur[2]; cur[2] = alt[2]; alt[2] = tmp; }

  // ---- block 2: mamba 3-5, self-attn 6-8, ffn 0-2 ----
  mamba_layer(cur[0], 3); mamba_layer(cur[1], 4); mamba_layer(cur[2], 5);
  sub(cur[0], cur[0], 6, cur[0]);
  sub(cur[1], cur[1], 7, cur[1]);
  sub(cur[2], cur[2], 8, cur[2]);
  ffn(cur[0], 0); ffn(cur[1], 1); ffn(cur[2], 2);

  // ---- final: torch_ln(t + v + a) -> d_out ----
  ln512_kernel<<<kR, 256, 0, stream>>>(cur[0], cur[1], cur[2],
                                       n1_g, n1_b, (float*)d_out, 1, kS);
}
